// SequenceClassifier_41790031790259
// MI455X (gfx1250) — compile-verified, matrix operand order checked by reading, not run on hardware
//
#include <hip/hip_runtime.h>
#include <hip/hip_bf16.h>
#include <math.h>

// Problem constants (from reference): B=128, T=2048, I=64, H=512
#define BB   128
#define TSEQ 2048
#define IN   64
#define HID  512
#define HP   520       // padded LDS row stride (bf16) -> conflict-free A-frag loads
#define XPF  16        // X prefetch distance (steps ahead)
#define SLAB (32 * 512) // bf16 elements per K-slice slab of packed W_hh

typedef __attribute__((ext_vector_type(16))) __bf16 v16bf;
typedef __attribute__((ext_vector_type(8)))  __bf16 v8bf;
typedef __attribute__((ext_vector_type(8)))  float  v8f;

// ---------------------------------------------------------------------------
// Fragment layout convention (mirrors ISA 16-bit A/B striping, 16x16x32 bf16):
//   lane l: n (or m) = l & 15, half = l >> 4
//   slot s in [0,16): k = 32*kt + (s&7) + ((s>>3)<<4) + (half<<3)
// => per lane the 16 bf16 are two contiguous 8-element runs (16B each).
// Packed weight buffer: frag f = kt*32 + nt, element addr = f*512 + lane*16 + s
// so each fragment is one contiguous, fully coalesced 1KB block, and the 4
// fragments a wave needs per K-slice are one contiguous 4KB block.
// ---------------------------------------------------------------------------

__global__ void rnn_pack_weights(const float* __restrict__ W_ih,  // [H][I]
                                 const float* __restrict__ W_hh,  // [H][H]
                                 const float* __restrict__ b_ih,  // [H]
                                 const float* __restrict__ b_hh,  // [H]
                                 __bf16* __restrict__ wpk_ih,     // [64*512]
                                 __bf16* __restrict__ wpk_hh,     // [512*512]
                                 float*  __restrict__ biasc)      // [H]
{
    int tid = blockIdx.x * blockDim.x + threadIdx.x;

    if (tid < HID) biasc[tid] = b_ih[tid] + b_hh[tid];

    // W_hh pack: B[k][n] = W_hh[n][k]   (h @ W_hh^T)
    if (tid < HID * HID) {
        int slot = tid & 15;
        int lane = (tid >> 4) & 31;
        int frag = tid >> 9;            // 0..511
        int kt   = frag >> 5;           // 0..15
        int nt   = frag & 31;           // 0..31
        int n = nt * 16 + (lane & 15);
        int k = kt * 32 + (slot & 7) + ((slot >> 3) << 4) + ((lane >> 4) << 3);
        wpk_hh[tid] = (__bf16)W_hh[n * HID + k];
    }
    // W_ih pack: B[k=i][n=h] = W_ih[h][i]
    if (tid < IN * HID) {
        int slot = tid & 15;
        int lane = (tid >> 4) & 31;
        int frag = tid >> 9;            // 0..63
        int nt   = frag & 31;
        int kt   = frag >> 5;           // 0..1
        int n = nt * 16 + (lane & 15);
        int k = kt * 32 + (slot & 7) + ((slot >> 3) << 4) + ((lane >> 4) << 3);
        wpk_ih[tid] = (__bf16)W_ih[n * IN + k];
    }
}

__device__ __forceinline__ v16bf join16(v8bf lo, v8bf hi) {
    return __builtin_shufflevector(lo, hi, 0, 1, 2, 3, 4, 5, 6, 7,
                                           8, 9, 10, 11, 12, 13, 14, 15);
}

__device__ __forceinline__ void wmma_bf16(v16bf a, v16bf b, v8f& c) {
    c = __builtin_amdgcn_wmma_f32_16x16x32_bf16(false, a, false, b,
                                                (short)0, c, false, false);
}

// Fast tanh: h is rounded to bf16 right after, so HW-transcendental accuracy
// (v_exp_f32 + v_rcp_f32, ~1 ulp each) is plenty. Saturates correctly at +-inf.
__device__ __forceinline__ float tanh_fast(float x) {
    float e = __expf(2.0f * x);
    return 1.0f - 2.0f * __builtin_amdgcn_rcpf(e + 1.0f);
}

// Persistent scan kernel: 8 workgroups x 256 threads (8 wave32, 2 waves/SIMD
// => per-wave VGPR cap ~256; ~190 live regs by design, no spills).
// Workgroup g owns batch rows [16g, 16g+16); wave w owns output cols [64w, 64w+64).
// Per step: acc = b_ih+b_hh; x-projection (register-stationary W_ih, 8 WMMA);
// recurrent GEMM (W_hh streamed from L2; K-loop unrolled x2 with ping-pong
// fragment sets => zero rotate-moves; 64 WMMA); fast tanh; h -> LDS (bf16).
__launch_bounds__(256)
__global__ void rnn_scan(const float*  __restrict__ X,        // [B][T][I]
                         const __bf16* __restrict__ wpk_ih,
                         const __bf16* __restrict__ wpk_hh,
                         const float*  __restrict__ biasc,    // [H] (b_ih+b_hh)
                         const float*  __restrict__ W_out,    // [1][H]
                         const float*  __restrict__ b_out,    // [1]
                         float*        __restrict__ out)      // [B][1]
{
    __shared__ __bf16 hlds[16 * HP];     // current hidden state tile, bf16

    const int lane = threadIdx.x & 31;
    const int wave = threadIdx.x >> 5;   // 0..7
    const int lm   = lane & 15;
    const int hf   = lane >> 4;
    const int b0   = blockIdx.x * 16;

    // h0 = 0
    for (int i = threadIdx.x; i < 16 * HP; i += 256) hlds[i] = (__bf16)0.0f;

    // Register-stationary W_ih fragments (2 K-slices x 4 N-tiles = 64 VGPRs)
    v16bf wihb[2][4];
#pragma unroll
    for (int kt = 0; kt < 2; ++kt)
#pragma unroll
        for (int ntl = 0; ntl < 4; ++ntl) {
            int nt = wave * 4 + ntl;
            wihb[kt][ntl] = *(const v16bf*)(wpk_ih + ((kt * 32 + nt) * 512 + lane * 16));
        }

    float biasv[4];
#pragma unroll
    for (int ntl = 0; ntl < 4; ++ntl) biasv[ntl] = biasc[wave * 64 + ntl * 16 + lm];

    // Per-wave base of its 4 consecutive W_hh fragments (frag = kt*32 + nt)
    const __bf16* wb    = wpk_hh + ((size_t)(wave * 4) * 512 + lane * 16);
    const float*  xbase = X + (size_t)(b0 + lm) * TSEQ * IN;

    __syncthreads();

    for (int t = 0; t < TSEQ; ++t) {
        // Prefetch X row XPF steps ahead (global_prefetch_b8: no VGPRs/counter).
        {
            int tp = t + XPF; if (tp > TSEQ - 1) tp = TSEQ - 1;
            __builtin_prefetch(xbase + (size_t)tp * IN, 0, 0);
        }

        // ---- issue step-top loads: X tile + first recurrent fragment set ----
        const float* xb = xbase + (size_t)t * IN;
        v8f x00 = *(const v8f*)(xb + hf * 8);
        v8f x01 = *(const v8f*)(xb + 16 + hf * 8);
        v8f x10 = *(const v8f*)(xb + 32 + hf * 8);
        v8f x11 = *(const v8f*)(xb + 48 + hf * 8);

        const __bf16* hrow = hlds + lm * HP + hf * 8;
        // ping-pong set P (even kt) / set Q (odd kt)
        v16bf aP = join16(*(const v8bf*)(hrow), *(const v8bf*)(hrow + 16));
        v16bf bP0 = *(const v16bf*)(wb);
        v16bf bP1 = *(const v16bf*)(wb + 512);
        v16bf bP2 = *(const v16bf*)(wb + 1024);
        v16bf bP3 = *(const v16bf*)(wb + 1536);

        v8f acc[4];
#pragma unroll
        for (int ntl = 0; ntl < 4; ++ntl)
#pragma unroll
            for (int v = 0; v < 8; ++v) acc[ntl][v] = biasv[ntl];

        // ---- input projection (recurrent preload latency hides here) ----
        v16bf a0, a1;
#pragma unroll
        for (int i = 0; i < 8; ++i) {
            a0[i] = (__bf16)x00[i]; a0[8 + i] = (__bf16)x01[i];
            a1[i] = (__bf16)x10[i]; a1[8 + i] = (__bf16)x11[i];
        }
#pragma unroll
        for (int ntl = 0; ntl < 4; ++ntl) wmma_bf16(a0, wihb[0][ntl], acc[ntl]);
#pragma unroll
        for (int ntl = 0; ntl < 4; ++ntl) wmma_bf16(a1, wihb[1][ntl], acc[ntl]);

        // ---- recurrent GEMM: K-loop unrolled x2, ping-pong sets, no moves ----
        const __bf16* wk = wb + SLAB;          // slab for kt = 2*kt2+1
#pragma unroll 1
        for (int kt2 = 0; kt2 < 7; ++kt2) {
            // load set Q (kt = 2*kt2+1)
            const __bf16* hq = hrow + (2 * kt2 + 1) * 32;
            v16bf aQ  = join16(*(const v8bf*)(hq), *(const v8bf*)(hq + 16));
            v16bf bQ0 = *(const v16bf*)(wk);
            v16bf bQ1 = *(const v16bf*)(wk + 512);
            v16bf bQ2 = *(const v16bf*)(wk + 1024);
            v16bf bQ3 = *(const v16bf*)(wk + 1536);
            // consume set P (kt = 2*kt2)
            wmma_bf16(aP, bP0, acc[0]);
            wmma_bf16(aP, bP1, acc[1]);
            wmma_bf16(aP, bP2, acc[2]);
            wmma_bf16(aP, bP3, acc[3]);
            // load set P (kt = 2*kt2+2)
            const __bf16* hp = hrow + (2 * kt2 + 2) * 32;
            aP  = join16(*(const v8bf*)(hp), *(const v8bf*)(hp + 16));
            bP0 = *(const v16bf*)(wk + SLAB);
            bP1 = *(const v16bf*)(wk + SLAB + 512);
            bP2 = *(const v16bf*)(wk + SLAB + 1024);
            bP3 = *(const v16bf*)(wk + SLAB + 1536);
            // consume set Q (kt = 2*kt2+1)
            wmma_bf16(aQ, bQ0, acc[0]);
            wmma_bf16(aQ, bQ1, acc[1]);
            wmma_bf16(aQ, bQ2, acc[2]);
            wmma_bf16(aQ, bQ3, acc[3]);
            wk += 2 * SLAB;
        }
        // kt = 14 is in set P; load kt = 15 into set Q, then drain both.
        {
            const __bf16* hq = hrow + 15 * 32;
            v16bf aQ  = join16(*(const v8bf*)(hq), *(const v8bf*)(hq + 16));
            v16bf bQ0 = *(const v16bf*)(wk);
            v16bf bQ1 = *(const v16bf*)(wk + 512);
            v16bf bQ2 = *(const v16bf*)(wk + 1024);
            v16bf bQ3 = *(const v16bf*)(wk + 1536);
            wmma_bf16(aP, bP0, acc[0]);
            wmma_bf16(aP, bP1, acc[1]);
            wmma_bf16(aP, bP2, acc[2]);
            wmma_bf16(aP, bP3, acc[3]);
            wmma_bf16(aQ, bQ0, acc[0]);
            wmma_bf16(aQ, bQ1, acc[1]);
            wmma_bf16(aQ, bQ2, acc[2]);
            wmma_bf16(aQ, bQ3, acc[3]);
        }

        __syncthreads();   // all waves done reading h_t
#pragma unroll
        for (int ntl = 0; ntl < 4; ++ntl) {
            int col = wave * 64 + ntl * 16 + lm;
#pragma unroll
            for (int v = 0; v < 8; ++v) {
                int m = v + 8 * hf;
                hlds[m * HP + col] = (__bf16)tanh_fast(acc[ntl][v]);
            }
        }
        __syncthreads();   // h_{t+1} visible to all waves
    }

    // logits + softmax over a size-1 axis (== 1.0 for finite logit)
    if (wave == 0 && lane < 16) {
        float z = 0.0f;
        for (int k = 0; k < HID; ++k) z += (float)hlds[lm * HP + k] * W_out[k];
        z += b_out[0];
        out[b0 + lm] = __expf(z - z);
    }
}

extern "C" void kernel_launch(void* const* d_in, const int* in_sizes, int n_in,
                              void* d_out, int out_size, void* d_ws, size_t ws_size,
                              hipStream_t stream) {
    const float* X     = (const float*)d_in[0];
    const float* W_ih  = (const float*)d_in[1];
    const float* b_ih  = (const float*)d_in[2];
    const float* W_hh  = (const float*)d_in[3];
    const float* b_hh  = (const float*)d_in[4];
    const float* W_out = (const float*)d_in[5];
    const float* b_out = (const float*)d_in[6];
    float* out = (float*)d_out;

    // Workspace layout: packed W_hh (512KB) | packed W_ih (64KB) | bias (2KB)
    char* ws = (char*)d_ws;
    __bf16* wpk_hh = (__bf16*)ws;                                  // 512*512*2 B
    __bf16* wpk_ih = (__bf16*)(ws + (size_t)HID * HID * 2);        // 64*512*2  B
    float*  biasc  = (float*) (ws + (size_t)HID * HID * 2 + (size_t)IN * HID * 2);

    // Pack weights (one thread per packed bf16 of the biggest buffer)
    rnn_pack_weights<<<(HID * HID + 255) / 256, 256, 0, stream>>>(
        W_ih, W_hh, b_ih, b_hh, wpk_ih, wpk_hh, biasc);

    // Persistent scan: 8 workgroups (one per 16-row batch tile), 8 waves each
    rnn_scan<<<BB / 16, 256, 0, stream>>>(X, wpk_ih, wpk_hh, biasc, W_out, b_out, out);
}
